// VectorChannel_90984587198690
// MI455X (gfx1250) — matrix operand order repaired and checked by reading, not compiled, not run on hardware
//
#include <hip/hip_runtime.h>
#include <hip/hip_bf16.h>

typedef __attribute__((ext_vector_type(16))) __bf16 v16bf;
typedef __attribute__((ext_vector_type(8)))  float  v8f;

#define LN_EPS 1e-5f

__device__ __forceinline__ unsigned short f2bf(float f) {
    unsigned u = __builtin_bit_cast(unsigned, f);
    unsigned r = u + 0x7FFFu + ((u >> 16) & 1u);   // round-to-nearest-even
    return (unsigned short)(r >> 16);
}

__device__ __forceinline__ float sigmoidf(float x) {
    return 1.0f / (1.0f + __expf(-x));
}

// ---------------------------------------------------------------------------
// Pack Wi [512, 256] (row-major, fp32) into bf16 WMMA B-fragments.
// Fragment (kt, nt): B is 32x16 (KxN). Lane L: n = nt*16 + (L&15),
// element j (0..15): k = kt*32 + 16*(L>>4) + j.
// Bpack[((kt*32 + nt)*32 + L)*16 + j]
// ---------------------------------------------------------------------------
__global__ void pack_wi_kernel(const float* __restrict__ Wi,
                               unsigned short* __restrict__ Bpack) {
    int idx = blockIdx.x * blockDim.x + threadIdx.x;     // 8*32*32*16 = 131072
    if (idx >= 8 * 32 * 32 * 16) return;
    int j    = idx & 15;
    int lane = (idx >> 4) & 31;
    int frag = idx >> 9;                  // kt*32 + nt
    int kt   = frag >> 5;
    int nt   = frag & 31;
    int k = kt * 32 + 16 * (lane >> 4) + j;
    int n = nt * 16 + (lane & 15);
    Bpack[idx] = f2bf(Wi[(size_t)n * 256 + k]);
}

// ---------------------------------------------------------------------------
// out = v_prev (seed for atomic scatter-add)
// ---------------------------------------------------------------------------
__global__ void copy_out_kernel(const float* __restrict__ src,
                                float* __restrict__ dst, int n) {
    int i = blockIdx.x * blockDim.x + threadIdx.x;
    if (i < n) dst[i] = src[i];
}

// ---------------------------------------------------------------------------
// vector_init: per node: SiLU(LN(h@W1^T+b1)) -> SiLU(LN(t@W2^T+b2)) -> [N,48]
// One wave32 per node. Lane handles outputs o=lane and o=lane+32 (lane<16).
// ---------------------------------------------------------------------------
__global__ void vector_init_kernel(const float* __restrict__ h,
                                   const float* __restrict__ W1, const float* __restrict__ b1,
                                   const float* __restrict__ g1, const float* __restrict__ bb1,
                                   const float* __restrict__ W2, const float* __restrict__ b2,
                                   const float* __restrict__ g2, const float* __restrict__ bb2,
                                   float* __restrict__ vectors, int N) {
    __shared__ float lds_t[8][48];
    int lane  = threadIdx.x & 31;
    int wslot = threadIdx.x >> 5;
    int wave  = blockIdx.x * 8 + wslot;
    bool valid = (wave < N);
    int node = valid ? wave : (N - 1);

    const float* hr = h + (size_t)node * 128;
    bool hi_lane = (lane < 16);

    // ---- layer 1: 128 -> 48 ----
    float x0 = b1[lane];
    float x1 = hi_lane ? b1[lane + 32] : 0.0f;
    const float* w0 = W1 + (size_t)lane * 128;
    const float* w1 = W1 + (size_t)(lane + (hi_lane ? 32 : 0)) * 128;
    for (int i = 0; i < 128; i += 4) {
        float4 hv = *(const float4*)(hr + i);
        float4 a  = *(const float4*)(w0 + i);
        x0 += hv.x * a.x + hv.y * a.y + hv.z * a.z + hv.w * a.w;
        if (hi_lane) {
            float4 c = *(const float4*)(w1 + i);
            x1 += hv.x * c.x + hv.y * c.y + hv.z * c.z + hv.w * c.w;
        }
    }
    // LN over 48 values striped across the wave
    float s  = x0 + (hi_lane ? x1 : 0.0f);
    float s2 = x0 * x0 + (hi_lane ? x1 * x1 : 0.0f);
    for (int d = 1; d < 32; d <<= 1) {
        s  += __shfl_xor(s, d, 32);
        s2 += __shfl_xor(s2, d, 32);
    }
    float mu  = s * (1.0f / 48.0f);
    float var = s2 * (1.0f / 48.0f) - mu * mu;
    float inv = rsqrtf(var + LN_EPS);
    float y0 = g1[lane] * (x0 - mu) * inv + bb1[lane];
    float t0 = y0 * sigmoidf(y0);
    lds_t[wslot][lane] = t0;
    if (hi_lane) {
        float y1 = g1[lane + 32] * (x1 - mu) * inv + bb1[lane + 32];
        lds_t[wslot][lane + 32] = y1 * sigmoidf(y1);
    }
    __syncthreads();

    // ---- layer 2: 48 -> 48 ----
    float z0 = b2[lane];
    float z1 = hi_lane ? b2[lane + 32] : 0.0f;
    const float* u0 = W2 + (size_t)lane * 48;
    const float* u1 = W2 + (size_t)(lane + (hi_lane ? 32 : 0)) * 48;
    const float* tv = lds_t[wslot];
    for (int i = 0; i < 48; i += 4) {
        float4 th = *(const float4*)(tv + i);
        float4 a  = *(const float4*)(u0 + i);
        z0 += th.x * a.x + th.y * a.y + th.z * a.z + th.w * a.w;
        if (hi_lane) {
            float4 c = *(const float4*)(u1 + i);
            z1 += th.x * c.x + th.y * c.y + th.z * c.z + th.w * c.w;
        }
    }
    s  = z0 + (hi_lane ? z1 : 0.0f);
    s2 = z0 * z0 + (hi_lane ? z1 * z1 : 0.0f);
    for (int d = 1; d < 32; d <<= 1) {
        s  += __shfl_xor(s, d, 32);
        s2 += __shfl_xor(s2, d, 32);
    }
    mu  = s * (1.0f / 48.0f);
    var = s2 * (1.0f / 48.0f) - mu * mu;
    inv = rsqrtf(var + LN_EPS);
    if (valid) {
        float y = g2[lane] * (z0 - mu) * inv + bb2[lane];
        vectors[(size_t)node * 48 + lane] = y * sigmoidf(y);
        if (hi_lane) {
            float yb = g2[lane + 32] * (z1 - mu) * inv + bb2[lane + 32];
            vectors[(size_t)node * 48 + lane + 32] = yb * sigmoidf(yb);
        }
    }
}

// ---------------------------------------------------------------------------
// Fused edge kernel: 16 edges per workgroup (256 threads = 8 waves).
//   A tile [16 x 256] bf16 in LDS (concat h[col], h[row])
//   GEMM via v_wmma_f32_16x16x32_bf16: each wave owns 64 columns
//   C [16 x 512] f32 in LDS (+bias), LN + sigmoid per row
//   stacked vectors [16 x 32 x 3] in LDS (vi + R_rel @ vj)
//   einsum contraction + atomicAdd scatter by col
// ---------------------------------------------------------------------------
__global__ void __launch_bounds__(256)
edge_kernel(const float* __restrict__ h,
            const long long* __restrict__ edge_index,   // [2, E]
            const float* __restrict__ frames,           // [N, 3, 3]
            const float* __restrict__ vectors,          // [N, 16, 3]
            const unsigned short* __restrict__ Bpack,   // packed Wi bf16
            const float* __restrict__ bi,
            const float* __restrict__ gi, const float* __restrict__ bbi,
            float* __restrict__ out, int N, int E) {
    __shared__ unsigned short ldsA[16 * 256];   // 8 KB  bf16 A tile
    __shared__ float ldsC[16 * 512];            // 32 KB f32 activations
    __shared__ float ldsSV[16 * 32 * 3];        // 6 KB  stacked vectors
    __shared__ int lds_row[16], lds_col[16];

    int tid = threadIdx.x;
    int e0  = blockIdx.x * 16;

    if (tid < 16) {
        int e = min(e0 + tid, E - 1);
        lds_row[tid] = (int)edge_index[e];       // sender
        lds_col[tid] = (int)edge_index[(size_t)E + e];  // receiver
    }
    __syncthreads();

    // ---- stage A tile: row m = edge, k<128 -> h[col], k>=128 -> h[row] ----
    {
        int m  = tid >> 4;
        int kb = (tid & 15) << 4;                // 16 floats per thread
        int node = (kb < 128) ? lds_col[m] : lds_row[m];
        const float4* src = (const float4*)(h + (size_t)node * 128 + (kb & 127));
        unsigned* dst = (unsigned*)(ldsA + m * 256 + kb);
#pragma unroll
        for (int t4 = 0; t4 < 4; ++t4) {
            float4 v = src[t4];
            dst[t4 * 2 + 0] = (unsigned)f2bf(v.x) | ((unsigned)f2bf(v.y) << 16);
            dst[t4 * 2 + 1] = (unsigned)f2bf(v.z) | ((unsigned)f2bf(v.w) << 16);
        }
    }
    __syncthreads();

    // ---- WMMA GEMM: [16 x 256] x [256 x 512] ----
    int lane = tid & 31;
    int wv   = tid >> 5;        // wave 0..7 -> columns [wv*64, wv*64+64)
    int half = lane >> 4;
    int lr   = lane & 15;

    v8f acc0 = {}, acc1 = {}, acc2 = {}, acc3 = {};
#pragma unroll
    for (int kt = 0; kt < 8; ++kt) {
        // A fragment: contiguous k = 16*half + s, same k order as Bpack
        union { unsigned u[8]; v16bf v; } af;
        const unsigned short* arow = ldsA + lr * 256 + kt * 32;
#pragma unroll
        for (int p = 0; p < 8; ++p) {
            int kloc = half * 16 + 2 * p;
            af.u[p] = *(const unsigned*)(arow + kloc);
        }
#pragma unroll
        for (int nt = 0; nt < 4; ++nt) {
            int ntabs = wv * 4 + nt;
            const v16bf bfr = *(const v16bf*)(Bpack +
                ((size_t)(kt * 32 + ntabs) * 32 + lane) * 16);
            v8f* accp = (nt == 0) ? &acc0 : (nt == 1) ? &acc1 : (nt == 2) ? &acc2 : &acc3;
            *accp = __builtin_amdgcn_wmma_f32_16x16x32_bf16(
                false, af.v, false, bfr, (short)0, *accp, false, false);
        }
    }

    // ---- C tile + bias -> LDS.  VGPR g: row = g + 8*half, col n = ntabs*16+lr
#pragma unroll
    for (int nt = 0; nt < 4; ++nt) {
        int n = (wv * 4 + nt) * 16 + lr;
        float bn = bi[n];
        v8f a = (nt == 0) ? acc0 : (nt == 1) ? acc1 : (nt == 2) ? acc2 : acc3;
#pragma unroll
        for (int g = 0; g < 8; ++g)
            ldsC[(g + 8 * half) * 512 + n] = a[g] + bn;
    }
    __syncthreads();

    // ---- LayerNorm(512) + sigmoid per edge row (16 threads per row) ----
    {
        int m = tid >> 4;
        int q = tid & 15;
        float s = 0.0f, s2 = 0.0f;
        for (int k = q; k < 512; k += 16) {
            float xv = ldsC[m * 512 + k];
            s += xv; s2 += xv * xv;
        }
        for (int d = 1; d < 16; d <<= 1) {
            s  += __shfl_xor(s, d, 16);
            s2 += __shfl_xor(s2, d, 16);
        }
        float mu  = s * (1.0f / 512.0f);
        float var = s2 * (1.0f / 512.0f) - mu * mu;
        float inv = rsqrtf(var + LN_EPS);
        for (int k = q; k < 512; k += 16) {
            float xv = ldsC[m * 512 + k];
            float y = gi[k] * (xv - mu) * inv + bbi[k];
            ldsC[m * 512 + k] = sigmoidf(y);
        }
    }

    // ---- stacked vectors: vi and R_rel @ vj  (thread = (edge m, channel c))
    {
        int m = tid >> 4;
        int c = tid & 15;
        int ci = lds_col[m], ri = lds_row[m];
        const float* Ri = frames + (size_t)ci * 9;
        const float* Rj = frames + (size_t)ri * 9;
        float Rr[3][3];
#pragma unroll
        for (int i = 0; i < 3; ++i)
#pragma unroll
            for (int k = 0; k < 3; ++k)
                Rr[i][k] = Ri[0 * 3 + i] * Rj[0 * 3 + k]
                         + Ri[1 * 3 + i] * Rj[1 * 3 + k]
                         + Ri[2 * 3 + i] * Rj[2 * 3 + k];
        const float* vip = vectors + (size_t)ci * 48 + c * 3;
        const float* vjp = vectors + (size_t)ri * 48 + c * 3;
        float vj0 = vjp[0], vj1 = vjp[1], vj2 = vjp[2];
#pragma unroll
        for (int o = 0; o < 3; ++o)
            ldsSV[(m * 32 + c) * 3 + o] = vip[o];
#pragma unroll
        for (int i = 0; i < 3; ++i)
            ldsSV[(m * 32 + 16 + c) * 3 + i] =
                Rr[i][0] * vj0 + Rr[i][1] * vj1 + Rr[i][2] * vj2;
    }
    __syncthreads();

    // ---- einsum: v_new[m,c,o] = sum_k SV[m,k,o] * w[m, k*16 + c]; scatter ----
    {
        int m = tid >> 4;
        int c = tid & 15;
        int e = e0 + m;
        if (e < E) {
            float a0 = 0.0f, a1 = 0.0f, a2 = 0.0f;
#pragma unroll
            for (int k = 0; k < 32; ++k) {
                float wgt = ldsC[m * 512 + k * 16 + c];
                const float* sv = &ldsSV[(m * 32 + k) * 3];
                a0 += sv[0] * wgt;
                a1 += sv[1] * wgt;
                a2 += sv[2] * wgt;
            }
            float* dst = out + (size_t)lds_col[m] * 48 + c * 3;
            atomicAdd(dst + 0, a0);
            atomicAdd(dst + 1, a1);
            atomicAdd(dst + 2, a2);
        }
    }
}

// ---------------------------------------------------------------------------
extern "C" void kernel_launch(void* const* d_in, const int* in_sizes, int n_in,
                              void* d_out, int out_size, void* d_ws, size_t ws_size,
                              hipStream_t stream) {
    const float*      h      = (const float*)d_in[1];
    const long long*  ei     = (const long long*)d_in[2];
    const float*      frames = (const float*)d_in[3];
    const float*      v_prev = (const float*)d_in[5];
    const float*      W1 = (const float*)d_in[6];
    const float*      b1 = (const float*)d_in[7];
    const float*      g1 = (const float*)d_in[8];
    const float*      bb1 = (const float*)d_in[9];
    const float*      W2 = (const float*)d_in[10];
    const float*      b2 = (const float*)d_in[11];
    const float*      g2 = (const float*)d_in[12];
    const float*      bb2 = (const float*)d_in[13];
    const float*      Wi = (const float*)d_in[14];
    const float*      bi = (const float*)d_in[15];
    const float*      gi = (const float*)d_in[16];
    const float*      bbi = (const float*)d_in[17];
    float* out = (float*)d_out;

    int N = in_sizes[0] / 3;        // x is [N,3]
    int E = in_sizes[2] / 2;        // edge_index is [2,E]

    // workspace: [ vectors: N*48 f32 | Bpack: 8*32*32*16 bf16 (256 KB) ]
    float* vectors = (float*)d_ws;
    size_t vecBytes = ((size_t)N * 48 * sizeof(float) + 255) & ~(size_t)255;
    unsigned short* Bpack = (unsigned short*)((char*)d_ws + vecBytes);

    pack_wi_kernel<<<(8 * 32 * 32 * 16 + 255) / 256, 256, 0, stream>>>(Wi, Bpack);
    copy_out_kernel<<<((N * 48) + 255) / 256, 256, 0, stream>>>(v_prev, out, N * 48);
    vector_init_kernel<<<(N + 7) / 8, 256, 0, stream>>>(
        h, W1, b1, g1, bb1, W2, b2, g2, bb2, vectors, N);
    edge_kernel<<<(E + 15) / 16, 256, 0, stream>>>(
        h, ei, frames, vectors, Bpack, bi, gi, bbi, out, N, E);
    (void)g2; (void)bb2; (void)n_in; (void)out_size; (void)ws_size;
}
